// GPTAttention_56530359550284
// MI455X (gfx1250) — matrix operand order repaired and checked by reading, not compiled, not run on hardware
//
#include <hip/hip_runtime.h>
#include <stdint.h>

// ---------------------------------------------------------------------------
// GPT attention forward for MI455X (gfx1250, wave32, WMMA).
// All matmuls run on v_wmma_f32_16x16x32_bf16.
// ---------------------------------------------------------------------------

typedef __attribute__((ext_vector_type(16))) __bf16 v16bf;
typedef __attribute__((ext_vector_type(8)))  float  v8f;
typedef __attribute__((ext_vector_type(4)))  float  f32x4;

#define NBATCH 2
#define NT     2048
#define NDMOD  1024
#define NH     16
#define HD     64
#define M_TOK  (NBATCH*NT)   // 4096
#define N_QKV  (3*NDMOD)     // 3072

union FragU { v16bf v; f32x4 h[2]; };

__device__ __forceinline__ unsigned short f2bf(float f) {
  unsigned u = __float_as_uint(f);
  u += 0x7FFFu + ((u >> 16) & 1u);   // round-to-nearest-even
  return (unsigned short)(u >> 16);
}

__device__ __forceinline__ float redmax16(float v) {
  v = fmaxf(v, __shfl_xor(v, 1, 32));
  v = fmaxf(v, __shfl_xor(v, 2, 32));
  v = fmaxf(v, __shfl_xor(v, 4, 32));
  v = fmaxf(v, __shfl_xor(v, 8, 32));
  return v;
}
__device__ __forceinline__ float redsum16(float v) {
  v += __shfl_xor(v, 1, 32);
  v += __shfl_xor(v, 2, 32);
  v += __shfl_xor(v, 4, 32);
  v += __shfl_xor(v, 8, 32);
  return v;
}

// --------------------------- conversion kernels ----------------------------

__global__ void cvt_bf16(const float* __restrict__ src,
                         unsigned short* __restrict__ dst, int n) {
  int i = blockIdx.x * blockDim.x + threadIdx.x;
  if (i < n) dst[i] = f2bf(src[i]);
}

// dst[c][r] = src[r][c]  (row-major [rows][cols] -> bf16 [cols][rows])
__global__ void cvt_bf16_T(const float* __restrict__ src,
                           unsigned short* __restrict__ dst,
                           int rows, int cols) {
  int i = blockIdx.x * blockDim.x + threadIdx.x;
  if (i < rows * cols) {
    int r = i / cols, c = i % cols;
    dst[(size_t)c * rows + r] = f2bf(src[i]);
  }
}

// ------------------------------ QKV GEMM -----------------------------------
// C[4096][3072] = X[4096][1024] @ Wqkv + bqkv, scattered into Q/K/Vt + present.
// Wave tile: 16(M) x 64(N). Block: 8 waves (2x4) => 32 x 256 tile.

__global__ __launch_bounds__(256, 2) void qkv_gemm(
    const unsigned short* __restrict__ Xbf,    // [4096][1024] bf16
    const unsigned short* __restrict__ WT,     // [3072][1024] bf16 (transposed)
    const float* __restrict__ bias,            // [3072]
    unsigned short* __restrict__ Qb,           // [B][H][T][64] bf16, *0.125
    unsigned short* __restrict__ Kb,           // [B][H][T][64] bf16
    unsigned short* __restrict__ Vt,           // [B][H][64][T] bf16
    float* __restrict__ present) {             // [B][2][H][T][64] f32
  const int lane = threadIdx.x & 31;
  const int wave = threadIdx.x >> 5;
  const int c = lane & 15, hlf = lane >> 4;
  const int m_base = blockIdx.x * 32 + (wave >> 2) * 16;
  const int n_base = blockIdx.y * 256 + (wave & 3) * 64;

  const unsigned short* arow  = Xbf + (size_t)(m_base + c) * NDMOD;
  const unsigned short* brow0 = WT  + (size_t)(n_base + c) * NDMOD;

  const v8f vzero = {0.f,0.f,0.f,0.f,0.f,0.f,0.f,0.f};
  v8f acc[4];
  #pragma unroll
  for (int t = 0; t < 4; ++t) acc[t] = vzero;

  for (int k0 = 0; k0 < NDMOD; k0 += 32) {
    FragU a;
    a.h[0] = *reinterpret_cast<const f32x4*>(arow + k0 + 8 * hlf);
    a.h[1] = *reinterpret_cast<const f32x4*>(arow + k0 + 16 + 8 * hlf);
    #pragma unroll
    for (int t = 0; t < 4; ++t) {
      const unsigned short* br = brow0 + (size_t)t * 16 * NDMOD;
      FragU b;
      b.h[0] = *reinterpret_cast<const f32x4*>(br + k0 + 8 * hlf);
      b.h[1] = *reinterpret_cast<const f32x4*>(br + k0 + 16 + 8 * hlf);
      acc[t] = __builtin_amdgcn_wmma_f32_16x16x32_bf16(
          false, a.v, false, b.v, (short)0, acc[t], false, false);
    }
  }

  // Epilogue: C-layout lane (c,hlf): row m = m_base + r + 8*hlf, col n.
  #pragma unroll
  for (int t = 0; t < 4; ++t) {
    #pragma unroll
    for (int r = 0; r < 8; ++r) {
      const int m = m_base + r + 8 * hlf;
      const int b = m >> 11, tok = m & (NT - 1);
      const int n = n_base + t * 16 + c;
      const float val = acc[t][r] + bias[n];
      const int region = n >> 10;          // 0=Q 1=K 2=V (wave-uniform)
      const int nr = n & (NDMOD - 1);
      const int head = nr >> 6, dd = nr & 63;
      const size_t bht = (((size_t)b * NH + head) * NT + tok) * HD + dd;
      if (region == 0) {
        Qb[bht] = f2bf(val * 0.125f);      // fold rsqrt(hd)
      } else if (region == 1) {
        Kb[bht] = f2bf(val);
        present[((((size_t)b * 2 + 0) * NH + head) * NT + tok) * HD + dd] = val;
      } else {
        Vt[(((size_t)b * NH + head) * HD + dd) * NT + tok] = f2bf(val);
        present[((((size_t)b * 2 + 1) * NH + head) * NT + tok) * HD + dd] = val;
      }
    }
  }
}

// --------------------------- flash attention -------------------------------
// One wave per 16-query tile; online softmax over 32-key chunks.
// S = Qtile(16x64) @ K^T   (2 wmma per 16-key tile, k=d)
// P routed through wave-private LDS (C-layout -> A-layout), then
// O += P(16x32) @ V(32x64) (4 wmma, n=d tiles).
// Strict causal mask (i > j), matching reference (diagonal masked, -1e10).

__global__ __launch_bounds__(256, 2) void attn(
    const unsigned short* __restrict__ Qb,     // [B][H][T][64]
    const unsigned short* __restrict__ Kb,     // [B][H][T][64]
    const unsigned short* __restrict__ Vt,     // [B][H][64][T]
    unsigned short* __restrict__ ctx) {        // [B][T][1024] bf16
  __shared__ unsigned short Plds[8][16][32];   // per-wave 16x32 P tile

  const int lane = threadIdx.x & 31;
  const int wave = threadIdx.x >> 5;
  const int c = lane & 15, hlf = lane >> 4;
  const int bh = blockIdx.x;                   // b*16 + head
  const int b = bh >> 4, head = bh & 15;
  const int q_tile = blockIdx.y * 8 + wave;
  const int q0 = q_tile * 16;

  // Q A-fragments (k0 = 0 and 32 of d=64), resident for whole kernel.
  const unsigned short* qrow = Qb + ((size_t)bh * NT + q0 + c) * HD;
  FragU aq0, aq1;
  aq0.h[0] = *reinterpret_cast<const f32x4*>(qrow + 8 * hlf);
  aq0.h[1] = *reinterpret_cast<const f32x4*>(qrow + 16 + 8 * hlf);
  aq1.h[0] = *reinterpret_cast<const f32x4*>(qrow + 32 + 8 * hlf);
  aq1.h[1] = *reinterpret_cast<const f32x4*>(qrow + 48 + 8 * hlf);

  const v8f vzero = {0.f,0.f,0.f,0.f,0.f,0.f,0.f,0.f};
  float rm[8], rl[8];
  #pragma unroll
  for (int r = 0; r < 8; ++r) { rm[r] = -3.0e38f; rl[r] = 0.f; }
  v8f O[4];
  #pragma unroll
  for (int t = 0; t < 4; ++t) O[t] = vzero;

  const int n_chunks = q_tile / 2 + 1;         // keys [0, 32*n_chunks) cover j<i
  for (int cc = 0; cc < n_chunks; ++cc) {
    const int key0 = cc * 32;

    // ---- scores: two 16x16 tiles over d=64 ----
    v8f S0 = vzero, S1 = vzero;
    {
      const unsigned short* kr0 = Kb + ((size_t)bh * NT + key0 + c) * HD;
      const unsigned short* kr1 = kr0 + 16 * HD;
      FragU bk;
      bk.h[0] = *reinterpret_cast<const f32x4*>(kr0 + 8 * hlf);
      bk.h[1] = *reinterpret_cast<const f32x4*>(kr0 + 16 + 8 * hlf);
      S0 = __builtin_amdgcn_wmma_f32_16x16x32_bf16(false, aq0.v, false, bk.v,
                                                   (short)0, S0, false, false);
      bk.h[0] = *reinterpret_cast<const f32x4*>(kr0 + 32 + 8 * hlf);
      bk.h[1] = *reinterpret_cast<const f32x4*>(kr0 + 48 + 8 * hlf);
      S0 = __builtin_amdgcn_wmma_f32_16x16x32_bf16(false, aq1.v, false, bk.v,
                                                   (short)0, S0, false, false);
      bk.h[0] = *reinterpret_cast<const f32x4*>(kr1 + 8 * hlf);
      bk.h[1] = *reinterpret_cast<const f32x4*>(kr1 + 16 + 8 * hlf);
      S1 = __builtin_amdgcn_wmma_f32_16x16x32_bf16(false, aq0.v, false, bk.v,
                                                   (short)0, S1, false, false);
      bk.h[0] = *reinterpret_cast<const f32x4*>(kr1 + 32 + 8 * hlf);
      bk.h[1] = *reinterpret_cast<const f32x4*>(kr1 + 48 + 8 * hlf);
      S1 = __builtin_amdgcn_wmma_f32_16x16x32_bf16(false, aq1.v, false, bk.v,
                                                   (short)0, S1, false, false);
    }

    // ---- masked online softmax (per physical row r + 8*hlf) ----
    float sc[8];
    #pragma unroll
    for (int r = 0; r < 8; ++r) {
      const int i  = q0 + r + 8 * hlf;
      const int j0 = key0 + c;
      const int j1 = key0 + 16 + c;
      float s0 = (i > j0) ? S0[r] : -1.0e10f;
      float s1 = (i > j1) ? S1[r] : -1.0e10f;
      float mx   = redmax16(fmaxf(s0, s1));
      float mnew = fmaxf(rm[r], mx);
      float scale = __expf(rm[r] - mnew);
      float p0 = __expf(s0 - mnew);
      float p1 = __expf(s1 - mnew);
      rl[r] = rl[r] * scale + redsum16(p0 + p1);
      rm[r] = mnew;
      sc[r] = scale;
      Plds[wave][r + 8 * hlf][c]      = f2bf(p0);
      Plds[wave][r + 8 * hlf][16 + c] = f2bf(p1);
    }
    #pragma unroll
    for (int t = 0; t < 4; ++t)
      #pragma unroll
      for (int r = 0; r < 8; ++r) O[t][r] *= sc[r];

    // ---- P (A-layout from LDS) @ V ----
    FragU pf;
    pf.h[0] = *reinterpret_cast<const f32x4*>(&Plds[wave][c][8 * hlf]);
    pf.h[1] = *reinterpret_cast<const f32x4*>(&Plds[wave][c][16 + 8 * hlf]);
    #pragma unroll
    for (int t = 0; t < 4; ++t) {
      const unsigned short* vr =
          Vt + (((size_t)bh * HD) + t * 16 + c) * NT + key0;
      FragU bv;
      bv.h[0] = *reinterpret_cast<const f32x4*>(vr + 8 * hlf);
      bv.h[1] = *reinterpret_cast<const f32x4*>(vr + 16 + 8 * hlf);
      O[t] = __builtin_amdgcn_wmma_f32_16x16x32_bf16(false, pf.v, false, bv.v,
                                                     (short)0, O[t], false, false);
    }
  }

  // ---- normalize + write ctx as [B][T][H*hd] (ready for projection) ----
  float inv[8];
  #pragma unroll
  for (int r = 0; r < 8; ++r) inv[r] = 1.0f / rl[r];
  #pragma unroll
  for (int t = 0; t < 4; ++t) {
    #pragma unroll
    for (int r = 0; r < 8; ++r) {
      const int tok = q0 + r + 8 * hlf;
      const int col = head * HD + t * 16 + c;
      ctx[((size_t)b * NT + tok) * NDMOD + col] = f2bf(O[t][r] * inv[r]);
    }
  }
}

// --------------------------- output projection -----------------------------

__global__ __launch_bounds__(256, 2) void proj_gemm(
    const unsigned short* __restrict__ Cb,     // [4096][1024] bf16 (ctx)
    const unsigned short* __restrict__ WT,     // [1024][1024] bf16 (transposed)
    const float* __restrict__ bias,            // [1024]
    float* __restrict__ out) {                 // [4096][1024] f32
  const int lane = threadIdx.x & 31;
  const int wave = threadIdx.x >> 5;
  const int c = lane & 15, hlf = lane >> 4;
  const int m_base = blockIdx.x * 32 + (wave >> 2) * 16;
  const int n_base = blockIdx.y * 256 + (wave & 3) * 64;

  const unsigned short* arow  = Cb + (size_t)(m_base + c) * NDMOD;
  const unsigned short* brow0 = WT + (size_t)(n_base + c) * NDMOD;

  const v8f vzero = {0.f,0.f,0.f,0.f,0.f,0.f,0.f,0.f};
  v8f acc[4];
  #pragma unroll
  for (int t = 0; t < 4; ++t) acc[t] = vzero;

  for (int k0 = 0; k0 < NDMOD; k0 += 32) {
    FragU a;
    a.h[0] = *reinterpret_cast<const f32x4*>(arow + k0 + 8 * hlf);
    a.h[1] = *reinterpret_cast<const f32x4*>(arow + k0 + 16 + 8 * hlf);
    #pragma unroll
    for (int t = 0; t < 4; ++t) {
      const unsigned short* br = brow0 + (size_t)t * 16 * NDMOD;
      FragU b;
      b.h[0] = *reinterpret_cast<const f32x4*>(br + k0 + 8 * hlf);
      b.h[1] = *reinterpret_cast<const f32x4*>(br + k0 + 16 + 8 * hlf);
      acc[t] = __builtin_amdgcn_wmma_f32_16x16x32_bf16(
          false, a.v, false, b.v, (short)0, acc[t], false, false);
    }
  }

  #pragma unroll
  for (int t = 0; t < 4; ++t) {
    #pragma unroll
    for (int r = 0; r < 8; ++r) {
      const int m = m_base + r + 8 * hlf;
      const int n = n_base + t * 16 + c;
      out[(size_t)m * NDMOD + n] = acc[t][r] + bias[n];
    }
  }
}

// ------------------------------- launcher ----------------------------------

extern "C" void kernel_launch(void* const* d_in, const int* in_sizes, int n_in,
                              void* d_out, int out_size, void* d_ws, size_t ws_size,
                              hipStream_t stream) {
  (void)in_sizes; (void)n_in; (void)out_size; (void)ws_size;

  const float* from  = (const float*)d_in[0];   // [2,2048,1024]
  const float* Wqkv  = (const float*)d_in[1];   // [1024,3072]
  const float* bqkv  = (const float*)d_in[2];   // [3072]
  const float* Wproj = (const float*)d_in[3];   // [1024,1024]
  const float* bproj = (const float*)d_in[4];   // [1024]

  float* out     = (float*)d_out;                       // [2,2048,1024]
  float* present = out + (size_t)M_TOK * NDMOD;         // [2,2,16,2048,64]

  // Workspace layout (bytes): 48 MB total.
  char* ws = (char*)d_ws;
  unsigned short* Xbf    = (unsigned short*)(ws + (size_t) 0 * 1024 * 1024);
  unsigned short* WqkvT  = (unsigned short*)(ws + (size_t) 8 * 1024 * 1024);
  unsigned short* WprojT = (unsigned short*)(ws + (size_t)14 * 1024 * 1024);
  unsigned short* Qb     = (unsigned short*)(ws + (size_t)16 * 1024 * 1024);
  unsigned short* Kbw    = (unsigned short*)(ws + (size_t)24 * 1024 * 1024);
  unsigned short* Vtw    = (unsigned short*)(ws + (size_t)32 * 1024 * 1024);
  unsigned short* ctxb   = (unsigned short*)(ws + (size_t)40 * 1024 * 1024);

  // 1) precision/layout conversions
  cvt_bf16 <<<(M_TOK * NDMOD + 255) / 256, 256, 0, stream>>>(from, Xbf, M_TOK * NDMOD);
  cvt_bf16_T<<<(NDMOD * N_QKV + 255) / 256, 256, 0, stream>>>(Wqkv, WqkvT, NDMOD, N_QKV);
  cvt_bf16_T<<<(NDMOD * NDMOD + 255) / 256, 256, 0, stream>>>(Wproj, WprojT, NDMOD, NDMOD);

  // 2) fused QKV projection (+ present output, + attention-friendly layouts)
  qkv_gemm<<<dim3(M_TOK / 32, N_QKV / 256), 256, 0, stream>>>(
      Xbf, WqkvT, bqkv, Qb, Kbw, Vtw, present);

  // 3) causal flash attention (strict mask i > j)
  attn<<<dim3(NBATCH * NH, NT / 128), 256, 0, stream>>>(Qb, Kbw, Vtw, ctxb);

  // 4) output projection
  proj_gemm<<<dim3(M_TOK / 32, NDMOD / 256), 256, 0, stream>>>(
      ctxb, WprojT, bproj, out);
}
